// DispCorrM_82214263980629
// MI455X (gfx1250) — matrix operand order repaired and checked by reading, not compiled, not run on hardware
//
#include <hip/hip_runtime.h>

// Disparity correlation (stereo cost volume) on MI455X / gfx1250.
// out[b,d,h,w] = (1/C) * sum_c L[b,c,h,w] * R[b,c,h,w-d], zero for w<d.
// Banded-GEMM mapping: per (b,h,w-tile) wave-task, 5 diagonal 16x16x32 tiles
// of L^T x R via v_wmma_f32_16x16x32_f16 (f16 in, f32 accumulate).
// 1/C = 2^-5 is folded exactly into the f32->f16 A conversion.

typedef __attribute__((ext_vector_type(16))) _Float16 v16h;
typedef __attribute__((ext_vector_type(8)))  float    v8f;

namespace {
constexpr int kB   = 4;
constexpr int kC2  = 64;
constexpr int kC   = 32;          // channels per side (K of the GEMM)
constexpr int kH   = 256;
constexpr int kW   = 512;
constexpr int kHW  = kH * kW;
constexpr int kWT  = kW / 16;     // 32 w-tiles per image row
constexpr float kInvC = 1.0f / 32.0f;   // 2^-5, exact in f16
}

// Fast path: fixed D == 64, NDELTA == 5, complementary-triangle store merge.
__global__ __launch_bounds__(256) void disp_corr_wmma_fast(
    const float* __restrict__ x, float* __restrict__ out, int ntasks)
{
    constexpr int kD = 64;
    const int task = (blockIdx.x << 3) + (threadIdx.x >> 5);
    if (task >= ntasks) return;            // uniform per wave

    const int i = task & (kWT - 1);
    const int h = (task >> 5) & (kH - 1);
    const int b = task >> 13;

    const int lane = threadIdx.x & 31;
    const int half = lane >> 4;            // lane group 0/1
    const int l15  = lane & 15;
    const int w0   = i << 4;
    const int M0   = half << 3;            // C/D row base for this lane group

    // ---- A fragment (16x32 f16): A[m,k] = L[b,k,h,w0+m] * 2^-5 ----
    // ISA 16-bit A 16x32 layout: lane group g: halfs 0..7 -> K=8g+j,
    // halfs 8..15 -> K=8g+16+(j-8).
    const int kbA = half << 3;
    const float* Abase = x + (((size_t)b * kC2) * kH + h) * (size_t)kW + (w0 + l15);
    v16h a;
#pragma unroll
    for (int j = 0; j < 16; ++j) {
        const int K = kbA + j + ((j >= 8) ? 8 : 0);
        a[j] = (_Float16)(Abase[(size_t)K * kHW] * kInvC);
    }

    // ---- 5 band tiles: B[k,n] = R[b,k,h,u0+n]; lane group g: K = 8g + j (+8 for j>=8) ----
    const int kbB = half << 3;
    const float* Rplane = x + (((size_t)b * kC2 + kC + kbB) * kH + h) * (size_t)kW;

    v8f acc[5];
#pragma unroll
    for (int dlt = 0; dlt < 5; ++dlt) {
        const int u0 = w0 - (dlt << 4);    // wave-uniform multiple of 16:
        if (u0 >= 0) {                     // valid for all lanes or none
            v16h bm;
#pragma unroll
            for (int j = 0; j < 16; ++j)   // unconditional, batched loads
                bm[j] = (_Float16)Rplane[(size_t)(j + ((j >= 8) ? 8 : 0)) * kHW + (u0 + l15)];
            v8f z = {};
            acc[dlt] = __builtin_amdgcn_wmma_f32_16x16x32_f16(
                false, a, false, bm, (short)0, z, false, false);
        } else {
            // Zero-pad region (w < d): product is zero, but d_out is poisoned
            // so zeros must still be stored below.
            v8f z = {};
            acc[dlt] = z;
        }
    }

    // ---- Stores. Element (r): d = 16*dlt + (M0 + r) - l15, w = w0 + M0 + r.
    // Offset of element r from element 0 of the same tile is the constant
    // r*(kHW+1) floats -> literal instruction offsets. Per-delta bases step by
    // 16*kHW floats. delta=0 keeps the M>=N triangle (d>=0), delta=4 keeps the
    // complementary M<N triangle (d<64): merge into unconditional select-stores.
    float* obase = out + (((size_t)b * kD) * kH + h) * (size_t)kW;
    float* p0 = obase + (long)(M0 - l15) * kHW + (w0 + M0);  // delta 0 base
    float* p1 = p0 + (size_t)16 * kHW;
    float* p2 = p0 + (size_t)32 * kHW;
    float* p3 = p0 + (size_t)48 * kHW;
    float* p4 = p0 + (size_t)64 * kHW;

#pragma unroll
    for (int r = 0; r < 8; ++r) {
        const size_t off = (size_t)r * (kHW + 1);
        p1[off] = acc[1][r];
        p2[off] = acc[2][r];
        p3[off] = acc[3][r];
        const bool sel = (M0 + r) >= l15;              // complementary triangles
        float* pe = sel ? p0 : p4;
        pe[off] = sel ? acc[0][r] : acc[4][r];
    }
}

// Generic fallback: runtime D / ndelta, full masks.
__global__ __launch_bounds__(256) void disp_corr_wmma_generic(
    const float* __restrict__ x, float* __restrict__ out,
    int D, int ndelta, int ntasks)
{
    const int task = (blockIdx.x << 3) + (threadIdx.x >> 5);
    if (task >= ntasks) return;

    const int i = task & (kWT - 1);
    const int h = (task >> 5) & (kH - 1);
    const int b = task >> 13;

    const int lane = threadIdx.x & 31;
    const int half = lane >> 4;
    const int l15  = lane & 15;
    const int w0   = i << 4;

    const int kbA = half << 3;
    const float* Abase = x + (((size_t)b * kC2) * kH + h) * (size_t)kW + (w0 + l15);
    v16h a;
#pragma unroll
    for (int j = 0; j < 16; ++j) {
        const int K = kbA + j + ((j >= 8) ? 8 : 0);
        a[j] = (_Float16)(Abase[(size_t)K * kHW] * kInvC);
    }

    const int kbB = half << 3;
    const float* Rplane = x + (((size_t)b * kC2 + kC + kbB) * kH + h) * (size_t)kW;
    float* obase = out + (((size_t)b * D) * kH + h) * (size_t)kW;

    for (int dlt = 0; dlt < ndelta; ++dlt) {
        const int u0 = w0 - (dlt << 4);
        v8f acc = {};
        if (u0 >= 0) {
            v16h bm;
#pragma unroll
            for (int j = 0; j < 16; ++j)
                bm[j] = (_Float16)Rplane[(size_t)(j + ((j >= 8) ? 8 : 0)) * kHW + (u0 + l15)];
            acc = __builtin_amdgcn_wmma_f32_16x16x32_f16(
                false, a, false, bm, (short)0, acc, false, false);
        }
#pragma unroll
        for (int r = 0; r < 8; ++r) {
            const int M = (half << 3) + r;
            const int d = (dlt << 4) + M - l15;
            if (d >= 0 && d < D)
                obase[(size_t)d * kHW + (w0 + M)] = acc[r];
        }
    }
}

extern "C" void kernel_launch(void* const* d_in, const int* in_sizes, int n_in,
                              void* d_out, int out_size, void* d_ws, size_t ws_size,
                              hipStream_t stream) {
    (void)in_sizes; (void)n_in; (void)d_ws; (void)ws_size;

    const float* x  = (const float*)d_in[0];
    float* out      = (float*)d_out;

    const int D      = out_size / (kB * kH * kW);     // 64 for this harness
    const int ndelta = (D + 15) / 16 + 1;
    const int ntasks = kB * kH * kWT;                 // 32768 wave-tasks
    const int nblocks = (ntasks + 7) / 8;             // 8 waves / 256-thread block

    if (D == 64 && ndelta == 5) {
        disp_corr_wmma_fast<<<dim3(nblocks), dim3(256), 0, stream>>>(
            x, out, ntasks);
    } else {
        disp_corr_wmma_generic<<<dim3(nblocks), dim3(256), 0, stream>>>(
            x, out, D, ndelta, ntasks);
    }
}
